// GCGRUCell_71665824301882
// MI455X (gfx1250) — matrix operand order repaired and checked by reading, not compiled, not run on hardware
//
#include <hip/hip_runtime.h>

#define B_      16
#define N_      2048
#define IN_DIM_ 128
#define U_      2
#define STEPS_  3
#define D_      130           // IN_DIM + U
#define DP_     160           // D padded to multiple of 32 (WMMA K-chunk)
#define M_      (B_ * N_)     // 32768 total rows (batch-major)

typedef __bf16 v16bf __attribute__((ext_vector_type(16)));
typedef float  v8f   __attribute__((ext_vector_type(8)));

union FragAB { unsigned u[8]; v16bf v; };

__device__ __forceinline__ unsigned short f2bf(float f) {
  unsigned u = __float_as_uint(f);
  u += 0x7fffu + ((u >> 16) & 1u);          // round-to-nearest-even
  return (unsigned short)(u >> 16);
}
__device__ __forceinline__ float sigmoidf_(float x) { return 1.0f / (1.0f + __expf(-x)); }

// ---------------------------------------------------------------------------
// prep: S = A0 + A1 (bf16), and S^T
// ---------------------------------------------------------------------------
__global__ void prep_S_kernel(const float* __restrict__ sup,
                              unsigned short* __restrict__ Sb,
                              unsigned short* __restrict__ STb) {
  size_t idx = (size_t)blockIdx.x * blockDim.x + threadIdx.x;
  if (idx >= (size_t)N_ * N_) return;
  int n = (int)(idx / N_), m = (int)(idx % N_);
  float s = sup[idx] + sup[(size_t)N_ * N_ + idx];
  unsigned short h = f2bf(s);
  Sb[idx] = h;
  STb[(size_t)m * N_ + n] = h;
}

// ---------------------------------------------------------------------------
// prep: W (3D x D) -> 3 TRANSPOSED zero-padded DP x DP bf16 chunks.
// WcT[chunk][n][k] = W[chunk*D + k][n]  so B fragments are K-contiguous.
// ---------------------------------------------------------------------------
__global__ void prep_W_kernel(const float* __restrict__ W, unsigned short* __restrict__ WcT) {
  int idx = blockIdx.x * blockDim.x + threadIdx.x;
  if (idx >= 3 * DP_ * DP_) return;
  int chunk = idx / (DP_ * DP_);
  int rem   = idx % (DP_ * DP_);
  int n = rem / DP_, k = rem % DP_;
  float v = (k < D_ && n < D_) ? W[(size_t)(chunk * D_ + k) * D_ + n] : 0.0f;
  WcT[idx] = f2bf(v);
}

// ---------------------------------------------------------------------------
// x0[m, d] = cat(inputs, state) zero-padded to DP
// ---------------------------------------------------------------------------
__global__ void init_x_kernel(const float* __restrict__ inp, const float* __restrict__ st,
                              float* __restrict__ X) {
  int idx = blockIdx.x * blockDim.x + threadIdx.x;
  if (idx >= M_ * DP_) return;
  int m = idx / DP_, d = idx % DP_;
  float v = 0.0f;
  if (d < IN_DIM_)      v = inp[(size_t)m * IN_DIM_ + d];
  else if (d < D_)      v = st[(size_t)m * U_ + (d - IN_DIM_)];
  X[idx] = v;
}

// ---------------------------------------------------------------------------
// cast x (fp32, [B][N][DP]) -> Xb bf16 row-major AND XbT bf16 column-major
// ([B][DP][N]) via LDS-tiled transpose. grid (N/32, DP/32, B), block (32, 8).
// ---------------------------------------------------------------------------
__global__ __launch_bounds__(256)
void cast_transpose_kernel(const float* __restrict__ X,
                           unsigned short* __restrict__ Xb,
                           unsigned short* __restrict__ XbT) {
  __shared__ float tile[32][33];
  const int b  = blockIdx.z;
  const int m0 = blockIdx.x * 32, d0 = blockIdx.y * 32;
  const int tx = threadIdx.x, ty = threadIdx.y;
#pragma unroll
  for (int i = 0; i < 4; ++i) {
    int mm = m0 + ty + i * 8;
    int dd = d0 + tx;
    float v = X[((size_t)b * N_ + mm) * DP_ + dd];
    tile[ty + i * 8][tx] = v;
    Xb[((size_t)b * N_ + mm) * DP_ + dd] = f2bf(v);
  }
  __syncthreads();
#pragma unroll
  for (int i = 0; i < 4; ++i) {
    int dd = d0 + ty + i * 8;
    int mm = m0 + tx;
    XbT[((size_t)b * DP_ + dd) * N_ + mm] = f2bf(tile[tx][ty + i * 8]);
  }
}

// ---------------------------------------------------------------------------
// Graph propagation GEMM:  Out[b] = S (N x N, bf16) @ X[b] (N x DP) -> bf16.
// No LDS, no barriers: A rows of S are K-contiguous; B columns come from the
// pre-transposed XbT so each lane's fragment is a contiguous 16-bf16 run
// (merges to global_load_b128 pairs). A/B use contiguous k = 16*(lane/16) + s order.
// grid = (N/128, DP/16, B), block = 256 (8 waves, each one 16x16 tile).
// ---------------------------------------------------------------------------
__global__ __launch_bounds__(256)
void graph_gemm_kernel(const unsigned short* __restrict__ Sb,
                       const unsigned short* __restrict__ XbT,   // [B][DP][N]
                       unsigned short* __restrict__ Out) {       // [B][N][DP]
  const int tid  = threadIdx.x;
  const int wave = tid >> 5, lane = tid & 31;
  const int lrow = lane & 15, hf = lane >> 4;
  const int d0   = blockIdx.y * 16;
  const int b    = blockIdx.z;
  const int m0   = blockIdx.x * 128 + wave * 16;

  const unsigned short* arow = Sb  + (size_t)(m0 + lrow) * N_;
  const unsigned short* brow = XbT + ((size_t)b * DP_ + d0 + lrow) * N_;

  v8f acc = {};
  for (int k0 = 0; k0 < N_; k0 += 32) {
    __builtin_prefetch(arow + k0 + 256, 0, 1);   // global_prefetch_b8, 512B ahead (speculative)
    __builtin_prefetch(brow + k0 + 256, 0, 1);
    FragAB a, bb;
    const unsigned* ap = (const unsigned*)(arow + k0);
    const unsigned* bp = (const unsigned*)(brow + k0);
#pragma unroll
    for (int v = 0; v < 8; ++v) {   // A: lanes 0-15 K0..15; lanes 16-31 K16..31
      int kk = 2 * v + 16 * hf;
      a.u[v] = ap[kk >> 1];
    }
#pragma unroll
    for (int v = 0; v < 8; ++v)     // B: col = lane&15; lanes<16 K0..15, lanes>=16 K16..31
      bb.u[v] = bp[8 * hf + v];
    acc = __builtin_amdgcn_wmma_f32_16x16x32_bf16(false, a.v, false, bb.v, (short)0, acc, false, false);
  }
#pragma unroll
  for (int v = 0; v < 8; ++v) {     // C/D: VGPR v -> row v (+8 upper lane half), col = lane&15
    int row = m0 + v + 8 * hf;
    Out[((size_t)b * N_ + row) * DP_ + d0 + lrow] = f2bf(acc[v]);
  }
}

// ---------------------------------------------------------------------------
// r/z GEMM: pre = a_in@W1 + a_out@W2 + x@W3 + b. Shared A fragments, two accs,
// B fragments straight from transposed weight chunks (L2-resident). No LDS.
// Epilogue: z (fp32) and bf16(r * x). grid = (M/128, DP/16), block = 256.
// ---------------------------------------------------------------------------
__global__ __launch_bounds__(256)
void rz_gemm_kernel(const unsigned short* __restrict__ Ain,
                    const unsigned short* __restrict__ Aout,
                    const unsigned short* __restrict__ Xb,
                    const unsigned short* __restrict__ WrT,  // 3 transposed DP x DP chunks
                    const unsigned short* __restrict__ WzT,
                    const float* __restrict__ br, const float* __restrict__ bz,
                    const float* __restrict__ Xf,
                    float* __restrict__ Zout,
                    unsigned short* __restrict__ RXb) {
  const int tid  = threadIdx.x;
  const int wave = tid >> 5, lane = tid & 31;
  const int lrow = lane & 15, hf = lane >> 4;
  const int d0   = blockIdx.y * 16;
  const int m0   = blockIdx.x * 128 + wave * 16;

  v8f accr = {}, accz = {};
#pragma unroll
  for (int src = 0; src < 3; ++src) {
    const unsigned short* A  = (src == 0) ? Ain : (src == 1) ? Aout : Xb;
    const unsigned short* arow = A + (size_t)(m0 + lrow) * DP_;
    const unsigned short* wr = WrT + (size_t)src * DP_ * DP_ + (size_t)(d0 + lrow) * DP_;
    const unsigned short* wz = WzT + (size_t)src * DP_ * DP_ + (size_t)(d0 + lrow) * DP_;
#pragma unroll
    for (int k0 = 0; k0 < DP_; k0 += 32) {
      FragAB a, b1, b2;
      const unsigned* ap = (const unsigned*)(arow + k0);
      const unsigned* p1 = (const unsigned*)(wr + k0);
      const unsigned* p2 = (const unsigned*)(wz + k0);
#pragma unroll
      for (int v = 0; v < 8; ++v) {
        int kk = 2 * v + 16 * hf;
        a.u[v] = ap[kk >> 1];
      }
#pragma unroll
      for (int v = 0; v < 8; ++v) { b1.u[v] = p1[8 * hf + v]; b2.u[v] = p2[8 * hf + v]; }
      accr = __builtin_amdgcn_wmma_f32_16x16x32_bf16(false, a.v, false, b1.v, (short)0, accr, false, false);
      accz = __builtin_amdgcn_wmma_f32_16x16x32_bf16(false, a.v, false, b2.v, (short)0, accz, false, false);
    }
  }
  const int col = d0 + lrow;
  const float biasr = (col < D_) ? br[col] : 0.0f;
  const float biasz = (col < D_) ? bz[col] : 0.0f;
#pragma unroll
  for (int v = 0; v < 8; ++v) {
    int row = m0 + v + 8 * hf;
    size_t idx = (size_t)row * DP_ + col;
    float r = sigmoidf_(accr[v] + biasr);
    float z = sigmoidf_(accz[v] + biasz);
    Zout[idx] = z;
    RXb[idx]  = f2bf(r * Xf[idx]);
  }
}

// ---------------------------------------------------------------------------
// h_hat GEMM + GRU update: x_next = (1-z)*x + z*tanh(a_in@W1 + a_out@W2 + rx@W3 + b)
// ---------------------------------------------------------------------------
__global__ __launch_bounds__(256)
void hh_gemm_kernel(const unsigned short* __restrict__ Ain,
                    const unsigned short* __restrict__ Aout,
                    const unsigned short* __restrict__ RXb,
                    const unsigned short* __restrict__ WhT,
                    const float* __restrict__ bh,
                    const float* __restrict__ Zbuf, const float* __restrict__ Xf,
                    float* __restrict__ Xnext) {
  const int tid  = threadIdx.x;
  const int wave = tid >> 5, lane = tid & 31;
  const int lrow = lane & 15, hf = lane >> 4;
  const int d0   = blockIdx.y * 16;
  const int m0   = blockIdx.x * 128 + wave * 16;

  v8f acc = {};
#pragma unroll
  for (int src = 0; src < 3; ++src) {
    const unsigned short* A  = (src == 0) ? Ain : (src == 1) ? Aout : RXb;
    const unsigned short* arow = A + (size_t)(m0 + lrow) * DP_;
    const unsigned short* wh = WhT + (size_t)src * DP_ * DP_ + (size_t)(d0 + lrow) * DP_;
#pragma unroll
    for (int k0 = 0; k0 < DP_; k0 += 32) {
      FragAB a, bb;
      const unsigned* ap = (const unsigned*)(arow + k0);
      const unsigned* wp = (const unsigned*)(wh + k0);
#pragma unroll
      for (int v = 0; v < 8; ++v) {
        int kk = 2 * v + 16 * hf;
        a.u[v] = ap[kk >> 1];
      }
#pragma unroll
      for (int v = 0; v < 8; ++v) bb.u[v] = wp[8 * hf + v];
      acc = __builtin_amdgcn_wmma_f32_16x16x32_bf16(false, a.v, false, bb.v, (short)0, acc, false, false);
    }
  }
  const int col = d0 + lrow;
  const float bias = (col < D_) ? bh[col] : 0.0f;
#pragma unroll
  for (int v = 0; v < 8; ++v) {
    int row = m0 + v + 8 * hf;
    size_t idx = (size_t)row * DP_ + col;
    float hhv = tanhf(acc[v] + bias);
    float z = Zbuf[idx];
    Xnext[idx] = (1.0f - z) * Xf[idx] + z * hhv;
  }
}

// ---------------------------------------------------------------------------
// gate head: value = sigmoid(x @ Wfc + bfc) (M x 4); st2 = value[:, :U] * state
// ---------------------------------------------------------------------------
__global__ void gate_fc_kernel(const float* __restrict__ Xf, const float* __restrict__ Wfc,
                               const float* __restrict__ bfc, const float* __restrict__ state,
                               float* __restrict__ val4, float* __restrict__ st2) {
  int idx = blockIdx.x * blockDim.x + threadIdx.x;
  if (idx >= M_ * 4) return;
  int m = idx >> 2, j = idx & 3;
  const float* xr = Xf + (size_t)m * DP_;
  float acc = bfc[j];
  for (int d = 0; d < D_; ++d) acc += xr[d] * Wfc[d * 4 + j];
  float v = sigmoidf_(acc);
  val4[idx] = v;
  if (j < U_) st2[m * U_ + j] = v * state[m * U_ + j];
}

// ---------------------------------------------------------------------------
// final: c = tanh(x_cand @ Wfc + bfc); out = u*state + (1-u)*c
// ---------------------------------------------------------------------------
__global__ void final_kernel(const float* __restrict__ Xf, const float* __restrict__ Wfc,
                             const float* __restrict__ bfc, const float* __restrict__ val4,
                             const float* __restrict__ state, float* __restrict__ out) {
  int idx = blockIdx.x * blockDim.x + threadIdx.x;
  if (idx >= M_ * U_) return;
  int m = idx >> 1, j = idx & 1;
  const float* xr = Xf + (size_t)m * DP_;
  float acc = bfc[j];
  for (int d = 0; d < D_; ++d) acc += xr[d] * Wfc[d * U_ + j];
  float c = tanhf(acc);
  float u = val4[m * 4 + U_ + j];
  out[idx] = u * state[idx] + (1.0f - u) * c;
}

// ---------------------------------------------------------------------------
extern "C" void kernel_launch(void* const* d_in, const int* in_sizes, int n_in,
                              void* d_out, int out_size, void* d_ws, size_t ws_size,
                              hipStream_t stream) {
  (void)in_sizes; (void)n_in; (void)out_size;
  const float* supports = (const float*)d_in[0];
  const float* inputs   = (const float*)d_in[1];
  const float* state    = (const float*)d_in[2];
  const float* gWr  = (const float*)d_in[3];
  const float* gWz  = (const float*)d_in[4];
  const float* gWh  = (const float*)d_in[5];
  const float* gbr  = (const float*)d_in[6];
  const float* gbz  = (const float*)d_in[7];
  const float* gbh  = (const float*)d_in[8];
  const float* gWfc = (const float*)d_in[9];
  const float* gbfc = (const float*)d_in[10];
  const float* cWr  = (const float*)d_in[11];
  const float* cWz  = (const float*)d_in[12];
  const float* cWh  = (const float*)d_in[13];
  const float* cbr  = (const float*)d_in[14];
  const float* cbz  = (const float*)d_in[15];
  const float* cbh  = (const float*)d_in[16];
  const float* cWfc = (const float*)d_in[17];
  const float* cbfc = (const float*)d_in[18];
  float* out = (float*)d_out;

  char* w = (char*)d_ws;
  auto take = [&](size_t bytes) -> char* {
    char* p = w; w += (bytes + 255) & ~((size_t)255); return p;
  };
  unsigned short* Sb    = (unsigned short*)take((size_t)N_ * N_ * 2);
  unsigned short* STb   = (unsigned short*)take((size_t)N_ * N_ * 2);
  unsigned short* xb    = (unsigned short*)take((size_t)M_ * DP_ * 2);
  unsigned short* xbT   = (unsigned short*)take((size_t)M_ * DP_ * 2);
  unsigned short* ainb  = (unsigned short*)take((size_t)M_ * DP_ * 2);
  unsigned short* aoutb = (unsigned short*)take((size_t)M_ * DP_ * 2);
  unsigned short* rxb   = (unsigned short*)take((size_t)M_ * DP_ * 2);
  float* x0 = (float*)take((size_t)M_ * DP_ * 4);
  float* x1 = (float*)take((size_t)M_ * DP_ * 4);
  float* zb = (float*)take((size_t)M_ * DP_ * 4);
  unsigned short* WgR = (unsigned short*)take((size_t)3 * DP_ * DP_ * 2);
  unsigned short* WgZ = (unsigned short*)take((size_t)3 * DP_ * DP_ * 2);
  unsigned short* WgH = (unsigned short*)take((size_t)3 * DP_ * DP_ * 2);
  unsigned short* WcR = (unsigned short*)take((size_t)3 * DP_ * DP_ * 2);
  unsigned short* WcZ = (unsigned short*)take((size_t)3 * DP_ * DP_ * 2);
  unsigned short* WcH = (unsigned short*)take((size_t)3 * DP_ * DP_ * 2);
  float* val4 = (float*)take((size_t)M_ * 4 * 4);
  float* st2  = (float*)take((size_t)M_ * U_ * 4);
  if ((size_t)(w - (char*)d_ws) > ws_size) return;  // workspace too small

  dim3 blk(256);
  prep_S_kernel<<<((unsigned)((size_t)N_ * N_ + 255) / 256), blk, 0, stream>>>(supports, Sb, STb);
  int wgrid = (3 * DP_ * DP_ + 255) / 256;
  prep_W_kernel<<<wgrid, blk, 0, stream>>>(gWr, WgR);
  prep_W_kernel<<<wgrid, blk, 0, stream>>>(gWz, WgZ);
  prep_W_kernel<<<wgrid, blk, 0, stream>>>(gWh, WgH);
  prep_W_kernel<<<wgrid, blk, 0, stream>>>(cWr, WcR);
  prep_W_kernel<<<wgrid, blk, 0, stream>>>(cWz, WcZ);
  prep_W_kernel<<<wgrid, blk, 0, stream>>>(cWh, WcH);

  const int eg = (M_ * DP_ + 255) / 256;
  const dim3 ct(N_ / 32, DP_ / 32, B_);
  const dim3 ctb(32, 8);
  const dim3 gg(N_ / 128, DP_ / 16, B_);
  const dim3 gr(M_ / 128, DP_ / 16);

  auto ggnn = [&](const float* st, const unsigned short* WR, const unsigned short* WZ,
                  const unsigned short* WH, const float* br, const float* bz,
                  const float* bh) -> float* {
    init_x_kernel<<<eg, blk, 0, stream>>>(inputs, st, x0);
    float* xc = x0; float* xn = x1;
    for (int s = 0; s < STEPS_; ++s) {
      cast_transpose_kernel<<<ct, ctb, 0, stream>>>(xc, xb, xbT);
      graph_gemm_kernel<<<gg, blk, 0, stream>>>(Sb,  xbT, ainb);
      graph_gemm_kernel<<<gg, blk, 0, stream>>>(STb, xbT, aoutb);
      rz_gemm_kernel<<<gr, blk, 0, stream>>>(ainb, aoutb, xb, WR, WZ, br, bz, xc, zb, rxb);
      hh_gemm_kernel<<<gr, blk, 0, stream>>>(ainb, aoutb, rxb, WH, bh, zb, xc, xn);
      float* t = xc; xc = xn; xn = t;
    }
    return xc;
  };

  float* xg = ggnn(state, WgR, WgZ, WgH, gbr, gbz, gbh);
  gate_fc_kernel<<<(M_ * 4 + 255) / 256, blk, 0, stream>>>(xg, gWfc, gbfc, state, val4, st2);
  float* xc2 = ggnn(st2, WcR, WcZ, WcH, cbr, cbz, cbh);
  final_kernel<<<(M_ * U_ + 255) / 256, blk, 0, stream>>>(xc2, cWfc, cbfc, val4, state, out);
}
